// HybridLHM_37168646979792
// MI455X (gfx1250) — matrix operand order repaired and checked by reading, not compiled, not run on hardware
//
#include <hip/hip_runtime.h>

#define BATCH   4
#define SEQ     2048
#define DMODEL  512
#define DINNER  1024
#define DSTATE  16
#define DCONV   4
#define NHEADS  4
#define HEADDIM 128   // DMODEL / NHEADS

typedef __attribute__((ext_vector_type(16))) __bf16 v16bf;
typedef __attribute__((ext_vector_type(8)))  float  v8f;

union BF16x16 { unsigned int u[8]; v16bf v; };

// native bf16 converts (RNE) -> hardware v_cvt, avoids manual bit-twiddle VALU chains
__device__ __forceinline__ unsigned pack2bf(float a, float b) {
  union { __bf16 h[2]; unsigned u; } p;
  p.h[0] = (__bf16)a;
  p.h[1] = (__bf16)b;
  return p.u;
}

__device__ __forceinline__ unsigned short f2bf(float f) {
  union { __bf16 h; unsigned short u; } p;
  p.h = (__bf16)f;
  return p.u;
}

__device__ __forceinline__ v8f zero8() {
  v8f z;
#pragma unroll
  for (int i = 0; i < 8; ++i) z[i] = 0.f;
  return z;
}

__device__ __forceinline__ float act_apply(float v, int act) {
  if (act == 1) return v / (1.f + __expf(-v));                      // silu
  if (act == 2) return 0.5f * v * (1.f + erff(v * 0.70710678f));    // gelu exact
  if (act == 3) return (v > 20.f) ? v : log1pf(__expf(v));          // softplus
  return v;
}

// ---------------------------------------------------------------------------
// Main GEMM (tile-exact): C[M,N] = epilogue(A[M,K] @ W[N,K]^T)
// REQUIRES: M % 64 == 0, N % 128 == 0, K % 32 == 0  (true for all model GEMMs)
// Block = 256 thr (8 waves). Block tile 64M x 128N x 32K.
// Wave grid 2(M) x 4(N); each wave computes 32x32 = 4 wmma / K-step.
// Staging: batched float4 loads -> native bf16 cvt -> 8B LDS stores; prefetch.
// ---------------------------------------------------------------------------
__global__ __launch_bounds__(256) void k_gemm(
    const float* __restrict__ A, const float* __restrict__ W,
    const float* __restrict__ bias, const float* res,
    float* C, int M, int N, int K, int act)
{
  __shared__ unsigned short As[64][40];    // 64 rows x 32 k (bf16), 8B-aligned rows
  __shared__ unsigned short Bs[128][40];   // 128 n-rows x 32 k (bf16)
  const int tid  = threadIdx.x;
  const int lane = tid & 31, wave = tid >> 5;
  const int wm   = wave & 1;     // 0..1 : 32-row macro tile
  const int wn   = wave >> 1;    // 0..3 : 32-col macro tile
  const int half = lane >> 4, l16 = lane & 15;
  const int bM = blockIdx.x * 64, bN = blockIdx.y * 128;

  v8f acc[2][2];
#pragma unroll
  for (int mi = 0; mi < 2; ++mi)
#pragma unroll
    for (int ni = 0; ni < 2; ++ni) acc[mi][ni] = zero8();

  for (int k0 = 0; k0 < K; k0 += 32) {
    const float* Ab = A + (size_t)bM * K + k0;
    const float* Wb = W + (size_t)bN * K + k0;

    // phase 1: issue ALL global loads (6 x b128 in flight per thread)
    float4 fa[2], fw[4];
#pragma unroll
    for (int it = 0; it < 2; ++it) {
      int idx = tid + it * 256;           // 0..511 : 64 rows x 8 float4
      fa[it] = *(const float4*)(Ab + (size_t)(idx >> 3) * K + (idx & 7) * 4);
    }
#pragma unroll
    for (int it = 0; it < 4; ++it) {
      int idx = tid + it * 256;           // 0..1023 : 128 rows x 8 float4
      fw[it] = *(const float4*)(Wb + (size_t)(idx >> 3) * K + (idx & 7) * 4);
    }
    // prefetch next K slab while this one computes
    if (k0 + 32 < K && tid < 128) {
      if (tid < 64) __builtin_prefetch(Ab + (size_t)tid * K + 32, 0, 1);
      __builtin_prefetch(Wb + (size_t)tid * K + 32, 0, 1);
    }
    // phase 2: convert + store to LDS
#pragma unroll
    for (int it = 0; it < 2; ++it) {
      int idx = tid + it * 256;
      *(uint2*)&As[idx >> 3][(idx & 7) * 4] =
          make_uint2(pack2bf(fa[it].x, fa[it].y), pack2bf(fa[it].z, fa[it].w));
    }
#pragma unroll
    for (int it = 0; it < 4; ++it) {
      int idx = tid + it * 256;
      *(uint2*)&Bs[idx >> 3][(idx & 7) * 4] =
          make_uint2(pack2bf(fw[it].x, fw[it].y), pack2bf(fw[it].z, fw[it].w));
    }
    __syncthreads();

    // A fragments (16x32, contiguous layout: lanes 0-15 K 0-15, 16-31 K 16-31)
    BF16x16 af[2], bfr[2];
#pragma unroll
    for (int v = 0; v < 8; ++v) {
      int ka = half * 16 + 2 * v;
      int kb = half * 16 + 2 * v;   // B frag: lane = N col, half selects K 0-15/16-31
#pragma unroll
      for (int mi = 0; mi < 2; ++mi) {
        const unsigned short* ar = As[wm * 32 + mi * 16 + l16];
        af[mi].u[v] = (unsigned)ar[ka] | ((unsigned)ar[ka + 1] << 16);
      }
#pragma unroll
      for (int ni = 0; ni < 2; ++ni) {
        const unsigned short* br = Bs[wn * 32 + ni * 16 + l16];
        bfr[ni].u[v] = (unsigned)br[kb] | ((unsigned)br[kb + 1] << 16);
      }
    }
#pragma unroll
    for (int mi = 0; mi < 2; ++mi)
#pragma unroll
      for (int ni = 0; ni < 2; ++ni)
        acc[mi][ni] = __builtin_amdgcn_wmma_f32_16x16x32_bf16(
            false, af[mi].v, false, bfr[ni].v, (short)0, acc[mi][ni], false, false);
    __syncthreads();
  }

#pragma unroll
  for (int mi = 0; mi < 2; ++mi) {
#pragma unroll
    for (int ni = 0; ni < 2; ++ni) {
      int col = bN + wn * 32 + ni * 16 + l16;
#pragma unroll
      for (int r = 0; r < 8; ++r) {
        int row = bM + wm * 32 + mi * 16 + half * 8 + r;   // C layout rows half*8+r
        float v = acc[mi][ni][r];
        if (bias) v += bias[col];
        v = act_apply(v, act);
        if (res) v += res[(size_t)row * N + col];
        C[(size_t)row * N + col] = v;
      }
    }
  }
}

// ---------------------------------------------------------------------------
// Narrow GEMM for B_w / C_w projections: C[M,16] = A[M,K] @ W[16,K]^T
// REQUIRES: M % 128 == 0, K % 32 == 0. Block 256 thr, tile 128M x 16N.
// ---------------------------------------------------------------------------
__global__ __launch_bounds__(256) void k_gemm_n16(
    const float* __restrict__ A, const float* __restrict__ W,
    float* __restrict__ C, int M, int K)
{
  __shared__ unsigned short As[128][40];
  __shared__ unsigned short Bs[16][40];
  const int tid  = threadIdx.x;
  const int lane = tid & 31, wave = tid >> 5;   // wave 0..7 -> 16-row tile
  const int half = lane >> 4, l16 = lane & 15;
  const int bM = blockIdx.x * 128;

  v8f acc = zero8();
  for (int k0 = 0; k0 < K; k0 += 32) {
    const float* Ab = A + (size_t)bM * K + k0;
    float4 fa[4], fw;
#pragma unroll
    for (int it = 0; it < 4; ++it) {
      int idx = tid + it * 256;           // 0..1023
      fa[it] = *(const float4*)(Ab + (size_t)(idx >> 3) * K + (idx & 7) * 4);
    }
    bool wload = tid < 128;
    if (wload)
      fw = *(const float4*)(W + (size_t)(tid >> 3) * K + k0 + (tid & 7) * 4);
#pragma unroll
    for (int it = 0; it < 4; ++it) {
      int idx = tid + it * 256;
      *(uint2*)&As[idx >> 3][(idx & 7) * 4] =
          make_uint2(pack2bf(fa[it].x, fa[it].y), pack2bf(fa[it].z, fa[it].w));
    }
    if (wload)
      *(uint2*)&Bs[tid >> 3][(tid & 7) * 4] =
          make_uint2(pack2bf(fw.x, fw.y), pack2bf(fw.z, fw.w));
    __syncthreads();
    BF16x16 af, bf;
#pragma unroll
    for (int v = 0; v < 8; ++v) {
      int ka = half * 16 + 2 * v;
      const unsigned short* ar = As[wave * 16 + l16];
      af.u[v] = (unsigned)ar[ka] | ((unsigned)ar[ka + 1] << 16);
      int kb = half * 16 + 2 * v;
      const unsigned short* br = Bs[l16];
      bf.u[v] = (unsigned)br[kb] | ((unsigned)br[kb + 1] << 16);
    }
    acc = __builtin_amdgcn_wmma_f32_16x16x32_bf16(false, af.v, false, bf.v,
                                                  (short)0, acc, false, false);
    __syncthreads();
  }
#pragma unroll
  for (int r = 0; r < 8; ++r) {
    int row = bM + wave * 16 + half * 8 + r;
    C[(size_t)row * DSTATE + l16] = acc[r];
  }
}

// ---------------------------------------------------------------------------
// Flash attention: one wave per 16-query tile per (b,h). hd = 128.
// Scores + PV via bf16 WMMA; online softmax with intra-half shuffles.
// qkv: [M,1536] (q|k|v), ctx out: [M,512]
// ---------------------------------------------------------------------------
__global__ __launch_bounds__(32) void k_attn(const float* __restrict__ qkv, float* ctx)
{
  __shared__ unsigned short Pl[16][34];
  const int wq  = blockIdx.x;
  const int b   = blockIdx.y / NHEADS;
  const int h   = blockIdx.y % NHEADS;
  const int lane = threadIdx.x & 31;
  const int half = lane >> 4, l16 = lane & 15;
  const float scale = 0.08838834764831845f;   // 1/sqrt(128)

  // Q: 4 A-fragments (16 queries x 32 dims each), pre-scaled
  BF16x16 qf[4];
  {
    const float* qp = qkv + ((size_t)(b * SEQ + wq * 16 + l16)) * 1536 + h * HEADDIM;
#pragma unroll
    for (int c = 0; c < 4; ++c) {
#pragma unroll
      for (int v = 0; v < 8; ++v) {
        int ka = c * 32 + half * 16 + 2 * v;
        qf[c].u[v] = pack2bf(qp[ka] * scale, qp[ka + 1] * scale);
      }
    }
  }

  v8f o[8];
#pragma unroll
  for (int n = 0; n < 8; ++n) o[n] = zero8();
  float mrow[8], lrow[8];
#pragma unroll
  for (int r = 0; r < 8; ++r) { mrow[r] = -1e30f; lrow[r] = 0.f; }

  for (int kt = 0; kt < SEQ / 32; ++kt) {
    // scores for 2 key tiles of 16
    v8f sc[2];
    sc[0] = zero8(); sc[1] = zero8();
#pragma unroll
    for (int t = 0; t < 2; ++t) {
      const float* kp = qkv + ((size_t)(b * SEQ + kt * 32 + t * 16 + l16)) * 1536
                        + DMODEL + h * HEADDIM;
#pragma unroll
      for (int c = 0; c < 4; ++c) {
        BF16x16 kf;   // B = K^T : 32 dims x 16 keys
#pragma unroll
        for (int v = 0; v < 8; ++v) {
          int d = c * 32 + half * 16 + 2 * v;
          kf.u[v] = pack2bf(kp[d], kp[d + 1]);
        }
        sc[t] = __builtin_amdgcn_wmma_f32_16x16x32_bf16(false, qf[c].v, false, kf.v,
                                                        (short)0, sc[t], false, false);
      }
    }
    // online softmax; row r lives in the 16 lanes of this half -> xor 1,2,4,8
    float scl[8];
#pragma unroll
    for (int r = 0; r < 8; ++r) {
      float mx = fmaxf(sc[0][r], sc[1][r]);
      mx = fmaxf(mx, __shfl_xor(mx, 1));
      mx = fmaxf(mx, __shfl_xor(mx, 2));
      mx = fmaxf(mx, __shfl_xor(mx, 4));
      mx = fmaxf(mx, __shfl_xor(mx, 8));
      float newm = fmaxf(mrow[r], mx);
      float rs   = __expf(mrow[r] - newm);
      mrow[r] = newm;
      float p0 = __expf(sc[0][r] - newm);
      float p1 = __expf(sc[1][r] - newm);
      float ps = p0 + p1;
      ps += __shfl_xor(ps, 1);
      ps += __shfl_xor(ps, 2);
      ps += __shfl_xor(ps, 4);
      ps += __shfl_xor(ps, 8);
      lrow[r] = lrow[r] * rs + ps;
      scl[r]  = rs;
      Pl[half * 8 + r][l16]      = f2bf(p0);   // C-layout -> row-major 16x32 in LDS
      Pl[half * 8 + r][16 + l16] = f2bf(p1);
    }
    __syncthreads();
#pragma unroll
    for (int n = 0; n < 8; ++n) {
#pragma unroll
      for (int r = 0; r < 8; ++r) o[n][r] *= scl[r];
    }
    // P as A-fragment (16 queries x 32 keys)
    BF16x16 pf;
#pragma unroll
    for (int v = 0; v < 8; ++v) {
      int ka = half * 16 + 2 * v;
      pf.u[v] = (unsigned)Pl[l16][ka] | ((unsigned)Pl[l16][ka + 1] << 16);
    }
    __syncthreads();
    // O += P @ V  (V B-fragments: 32 keys x 16 dims per output tile)
#pragma unroll
    for (int n = 0; n < 8; ++n) {
      BF16x16 vf;
#pragma unroll
      for (int v = 0; v < 8; ++v) {
        const float* vp = qkv + ((size_t)(b * SEQ + kt * 32 + half * 16 + 2 * v)) * 1536
                          + 2 * DMODEL + h * HEADDIM + n * 16 + l16;
        vf.u[v] = pack2bf(vp[0], vp[1536]);
      }
      o[n] = __builtin_amdgcn_wmma_f32_16x16x32_bf16(false, pf.v, false, vf.v,
                                                     (short)0, o[n], false, false);
    }
  }
#pragma unroll
  for (int n = 0; n < 8; ++n) {
#pragma unroll
    for (int r = 0; r < 8; ++r) {
      int row = wq * 16 + half * 8 + r;
      ctx[((size_t)(b * SEQ + row)) * DMODEL + h * HEADDIM + n * 16 + l16] =
          o[n][r] / lrow[r];
    }
  }
}

// ---------------------------------------------------------------------------
// LayerNorm (one token per block)
// ---------------------------------------------------------------------------
__global__ __launch_bounds__(256) void k_ln(const float* __restrict__ x,
                                            const float* __restrict__ g,
                                            const float* __restrict__ bta,
                                            float* __restrict__ y, int D)
{
  __shared__ float s1[256], s2[256];
  const int m = blockIdx.x;
  const float* xr = x + (size_t)m * D;
  float a = 0.f, q = 0.f;
  for (int d = threadIdx.x; d < D; d += 256) { float v = xr[d]; a += v; q += v * v; }
  s1[threadIdx.x] = a; s2[threadIdx.x] = q;
  __syncthreads();
  for (int o = 128; o > 0; o >>= 1) {
    if ((int)threadIdx.x < o) {
      s1[threadIdx.x] += s1[threadIdx.x + o];
      s2[threadIdx.x] += s2[threadIdx.x + o];
    }
    __syncthreads();
  }
  float mu  = s1[0] / D;
  float inv = rsqrtf(s2[0] / D - mu * mu + 1e-5f);
  float* yr = y + (size_t)m * D;
  for (int d = threadIdx.x; d < D; d += 256)
    yr[d] = (xr[d] - mu) * inv * g[d] + bta[d];
}

// ---------------------------------------------------------------------------
// Depthwise causal conv (width 4) + bias + silu.  in: xz[:, :1024] of [M,2048]
// ---------------------------------------------------------------------------
__global__ void k_conv(const float* __restrict__ xz, const float* __restrict__ cw,
                       const float* __restrict__ cb, float* __restrict__ out, int M)
{
  int i = blockIdx.x * 256 + threadIdx.x;
  if (i >= M * DINNER) return;
  int m = i / DINNER, c = i % DINNER;
  int b = m / SEQ, t = m % SEQ;
  float acc = cb[c];
#pragma unroll
  for (int j = 0; j < DCONV; ++j) {
    int tt = t - (DCONV - 1) + j;
    if (tt >= 0)
      acc += xz[((size_t)b * SEQ + tt) * (2 * DINNER) + c] * cw[c * DCONV + j];
  }
  out[i] = acc / (1.f + __expf(-acc));   // silu
}

// ---------------------------------------------------------------------------
// Selective-scan: 16 lanes = 16 SSM states per channel; y reduced via shfl_xor.
// Fused: y = (scan + xc*D) * silu(z).  grid (DINNER/16, BATCH), block 256.
// ---------------------------------------------------------------------------
__global__ __launch_bounds__(256) void k_scan(
    const float* __restrict__ dt, const float* __restrict__ xc,
    const float* __restrict__ Bm, const float* __restrict__ Cm,
    const float* __restrict__ Alog, const float* __restrict__ Dp,
    const float* __restrict__ xz, float* __restrict__ y)
{
  const int b    = blockIdx.y;
  const int chan = blockIdx.x * 16 + (threadIdx.x >> 4);
  const int s    = threadIdx.x & 15;
  const float A  = -__expf(Alog[chan * DSTATE + s]);
  const float Dc = Dp[chan];
  float h = 0.f;
  for (int t = 0; t < SEQ; ++t) {
    size_t m = (size_t)b * SEQ + t;
    float dtv = dt[m * DINNER + chan];
    float xv  = xc[m * DINNER + chan];
    float bv  = Bm[m * DSTATE + s];
    float cv  = Cm[m * DSTATE + s];
    h = h * __expf(A * dtv) + xv * dtv * bv;
    float part = h * cv;
    part += __shfl_xor(part, 1);
    part += __shfl_xor(part, 2);
    part += __shfl_xor(part, 4);
    part += __shfl_xor(part, 8);
    if (s == 0) {
      float z = xz[m * (2 * DINNER) + DINNER + chan];
      y[m * DINNER + chan] = (part + xv * Dc) * (z / (1.f + __expf(-z)));
    }
  }
}

// ---------------------------------------------------------------------------
// Time encoding: enc = [sin(t*f+p), cos(t*f+p)]
// ---------------------------------------------------------------------------
__global__ void k_timeenc(const float* __restrict__ ts, const float* __restrict__ fr,
                          const float* __restrict__ ph, float* __restrict__ enc, int M)
{
  int i = blockIdx.x * 256 + threadIdx.x;
  if (i >= M * (DMODEL / 2)) return;
  int m = i >> 8, j = i & 255;
  float a = ts[m] * fr[j] + ph[j];
  enc[(size_t)m * DMODEL + j]       = sinf(a);
  enc[(size_t)m * DMODEL + 256 + j] = cosf(a);
}

__global__ void k_embed(const int* __restrict__ ids, const int* __restrict__ tt,
                        const float* __restrict__ emb, const float* __restrict__ temb,
                        float* x, int M)
{
  int i = blockIdx.x * 256 + threadIdx.x;
  if (i >= M * DMODEL) return;
  int m = i / DMODEL, d = i % DMODEL;
  x[i] += emb[(size_t)ids[m] * DMODEL + d] + temb[(size_t)tt[m] * DMODEL + d];
}

// ---------------------------------------------------------------------------
// Heads: LN(cls) @ r_w + r_bias ; LN(cls) @ m_w + m_bias. One block per batch.
// ---------------------------------------------------------------------------
__global__ __launch_bounds__(256) void k_heads(const float* __restrict__ x,
    const float* rg, const float* rb, const float* rw, const float* rbias,
    const float* mg, const float* mb, const float* mw, const float* mbias,
    float* out)
{
  __shared__ float s1[256], s2[256];
  const int b = blockIdx.x;
  const float* xr = x + (size_t)b * SEQ * DMODEL;   // token 0 = cls
  float a = 0.f, q = 0.f;
  for (int d = threadIdx.x; d < DMODEL; d += 256) { float v = xr[d]; a += v; q += v * v; }
  s1[threadIdx.x] = a; s2[threadIdx.x] = q;
  __syncthreads();
  for (int o = 128; o > 0; o >>= 1) {
    if ((int)threadIdx.x < o) {
      s1[threadIdx.x] += s1[threadIdx.x + o];
      s2[threadIdx.x] += s2[threadIdx.x + o];
    }
    __syncthreads();
  }
  float mu  = s1[0] / DMODEL;
  float inv = rsqrtf(s2[0] / DMODEL - mu * mu + 1e-5f);
  __syncthreads();
  float pr = 0.f, pm = 0.f;
  for (int d = threadIdx.x; d < DMODEL; d += 256) {
    float nv = (xr[d] - mu) * inv;
    pr += (nv * rg[d] + rb[d]) * rw[d];
    pm += (nv * mg[d] + mb[d]) * mw[d];
  }
  s1[threadIdx.x] = pr; s2[threadIdx.x] = pm;
  __syncthreads();
  for (int o = 128; o > 0; o >>= 1) {
    if ((int)threadIdx.x < o) {
      s1[threadIdx.x] += s1[threadIdx.x + o];
      s2[threadIdx.x] += s2[threadIdx.x + o];
    }
    __syncthreads();
  }
  if (threadIdx.x == 0) {
    out[b]         = s1[0] + rbias[0];
    out[BATCH + b] = s2[0] + mbias[0];
  }
}

// ---------------------------------------------------------------------------
static inline void launch_gemm(hipStream_t s, const float* A, const float* W,
                               const float* bias, const float* res, float* C,
                               int M, int N, int K, int act)
{
  dim3 g(M / 64, N / 128);
  k_gemm<<<g, 256, 0, s>>>(A, W, bias, res, C, M, N, K, act);
}

extern "C" void kernel_launch(void* const* d_in, const int* in_sizes, int n_in,
                              void* d_out, int out_size, void* d_ws, size_t ws_size,
                              hipStream_t stream)
{
  (void)in_sizes; (void)n_in; (void)out_size; (void)ws_size;
  const int* ids = (const int*)d_in[0];
  const int* tts = (const int*)d_in[1];
  const float* ts = (const float*)d_in[2];
  // params flattened in dict order:
  // 3 emb, 4 type_emb, 5 t_freqs, 6 t_phases, 7 t_w, 8 t_b,
  // 9.. blocks (8 x 12), 105 r_g 106 r_b 107 r_w 108 r_bias 109 m_g 110 m_b 111 m_w 112 m_bias
  auto F = [&](int i) { return (const float*)d_in[i]; };
  const int M = BATCH * SEQ;

  float* ws   = (float*)d_ws;
  float* xbuf = ws;  ws += (size_t)M * DMODEL;      // hidden state
  float* sbuf = ws;  ws += (size_t)M * DMODEL;      // LN output
  float* t0   = ws;  ws += (size_t)M * 2048;        // xz / qkv / mlp hidden
  float* t1   = ws;  ws += (size_t)M * DINNER;      // xc / enc
  float* t2   = ws;  ws += (size_t)M * DINNER;      // dt
  float* t3   = ws;  ws += (size_t)M * DINNER;      // y / attn ctx
  float* bmb  = ws;  ws += (size_t)M * DSTATE;      // Bm
  float* cmb  = ws;  ws += (size_t)M * DSTATE;      // Cm

  // Embedding: time encoding -> GEMM -> add token/type embeddings
  {
    int tot = M * (DMODEL / 2);
    k_timeenc<<<(tot + 255) / 256, 256, 0, stream>>>(ts, F(5), F(6), t1, M);
  }
  launch_gemm(stream, t1, F(7), F(8), nullptr, xbuf, M, DMODEL, DMODEL, 0);
  {
    int tot = M * DMODEL;
    k_embed<<<(tot + 255) / 256, 256, 0, stream>>>(ids, tts, F(3), F(4), xbuf, M);
  }

  for (int blk = 0; blk < 8; ++blk) {
    int p = 9 + blk * 12;
    bool is_attn = (blk == 3) || (blk == 7);
    if (!is_attn) {
      // p: norm_g, norm_b, in_w, conv_w, conv_b, dt_w, dt_b, B_w, C_w, D, A_log, out_w
      k_ln<<<M, 256, 0, stream>>>(xbuf, F(p + 0), F(p + 1), sbuf, DMODEL);
      launch_gemm(stream, sbuf, F(p + 2), nullptr, nullptr, t0, M, 2 * DINNER, DMODEL, 0);
      {
        int tot = M * DINNER;
        k_conv<<<(tot + 255) / 256, 256, 0, stream>>>(t0, F(p + 3), F(p + 4), t1, M);
      }
      launch_gemm(stream, t1, F(p + 5), F(p + 6), nullptr, t2, M, DINNER, DINNER, 3); // softplus
      k_gemm_n16<<<M / 128, 256, 0, stream>>>(t1, F(p + 7), bmb, M, DINNER);
      k_gemm_n16<<<M / 128, 256, 0, stream>>>(t1, F(p + 8), cmb, M, DINNER);
      {
        dim3 g(DINNER / 16, BATCH);
        k_scan<<<g, 256, 0, stream>>>(t2, t1, bmb, cmb, F(p + 10), F(p + 9), t0, t3);
      }
      launch_gemm(stream, t3, F(p + 11), nullptr, xbuf, xbuf, M, DMODEL, DINNER, 0);
    } else {
      // p: norm1_g, norm1_b, in_w, in_b, out_w, out_b, norm2_g, norm2_b, w1, b1, w2, b2
      k_ln<<<M, 256, 0, stream>>>(xbuf, F(p + 0), F(p + 1), sbuf, DMODEL);
      launch_gemm(stream, sbuf, F(p + 2), F(p + 3), nullptr, t0, M, 3 * DMODEL, DMODEL, 0);
      {
        dim3 g(SEQ / 16, BATCH * NHEADS);
        k_attn<<<g, 32, 0, stream>>>(t0, t3);
      }
      launch_gemm(stream, t3, F(p + 4), F(p + 5), xbuf, xbuf, M, DMODEL, DMODEL, 0);
      k_ln<<<M, 256, 0, stream>>>(xbuf, F(p + 6), F(p + 7), sbuf, DMODEL);
      launch_gemm(stream, sbuf, F(p + 8), F(p + 9), nullptr, t0, M, 4 * DMODEL, DMODEL, 2); // gelu
      launch_gemm(stream, t0, F(p + 10), F(p + 11), xbuf, xbuf, M, DMODEL, 4 * DMODEL, 0);
    }
  }

  k_heads<<<BATCH, 256, 0, stream>>>(xbuf, F(105), F(106), F(107), F(108),
                                     F(109), F(110), F(111), F(112), (float*)d_out);
}
